// MultiHeadClassifier_8169027797034
// MI455X (gfx1250) — compile-verified, matrix operand order checked by reading, not run on hardware
//
#include <hip/hip_runtime.h>
#include <hip/hip_bf16.h>

// ---------------------------------------------------------------------------
// Multi-head classifier: task bucketing + bf16x3-split WMMA grouped GEMM.
// B=65536, D=1024, T=16, H=128, C=10.
// - W1 pre-packed per launch into WMMA-fragment-swizzled bf16 hi/lo arrays
//   (inner loop is pure ds_load_b128 / global_load_b128 / v_wmma).
// - x staging software-pipelined (register prefetch of next K chunk).
// - x loads + out stores are non-temporal so the 256MB x stream doesn't
//   evict the L2-resident packed W1.
// ---------------------------------------------------------------------------

#define BN 65536
#define DN 1024
#define TN 16
#define HN 128
#define CN 10

#define MTILE 32                       // rows per workgroup
#define BUCK_CAP (BN + TN * MTILE)     // padded bucket capacity
#define NTILES   (BUCK_CAP / MTILE)    // 2064

// int workspace layout:
#define WS_COUNT 0
#define WS_OFF   16    // 17 entries, multiples of 32
#define WS_CUR   48
#define WS_BUCK  64
#define WS_INTS  (WS_BUCK + BUCK_CAP)

// packed W1: [t][ks(32)][cb(8)][lane(32)][j(16)] bf16 ; hi then lo array
#define WPACK_ELEMS (TN * 32 * 8 * 32 * 16)   // 2,097,152 per array

typedef __bf16 bf16x16 __attribute__((ext_vector_type(16)));
typedef float  f32x8   __attribute__((ext_vector_type(8)));
typedef float  f32x4   __attribute__((ext_vector_type(4)));

// element j of a v16bf A/B fragment -> K offset within a 32-wide K step
// (CDNA5 16-bit 16x32 layout: lanes 0-15 hold K 0-7,16-23; lanes 16-31
// hold K 8-15,24-31; 2 values per VGPR).
__device__ __forceinline__ int kmap(int j, int half) {
    return ((j < 8) ? j : j + 8) + (half << 3);
}

// ---------------------------------------------------------------------------
__global__ void mhc_init(int* ws) {
    int t = threadIdx.x;
    if (t < TN) { ws[WS_COUNT + t] = 0; ws[WS_CUR + t] = 0; }
}

__global__ void mhc_hist(const int* __restrict__ task, int* ws) {
    int i = blockIdx.x * blockDim.x + threadIdx.x;
    if (i < BN) atomicAdd(&ws[WS_COUNT + task[i]], 1);
}

__global__ void mhc_offsets(int* ws) {
    if (threadIdx.x == 0 && blockIdx.x == 0) {
        int off = 0;
        for (int t = 0; t < TN; ++t) {
            ws[WS_OFF + t] = off;
            off += ((ws[WS_COUNT + t] + MTILE - 1) / MTILE) * MTILE;
        }
        ws[WS_OFF + TN] = off;
    }
}

__global__ void mhc_fill(int* ws) {
    int i = blockIdx.x * blockDim.x + threadIdx.x;
    if (i < BUCK_CAP) ws[WS_BUCK + i] = -1;
}

__global__ void mhc_scatter(const int* __restrict__ task, int* ws) {
    int i = blockIdx.x * blockDim.x + threadIdx.x;
    if (i < BN) {
        int t = task[i];
        int pos = ws[WS_OFF + t] + atomicAdd(&ws[WS_CUR + t], 1);
        ws[WS_BUCK + pos] = i;
    }
}

// ---------------------------------------------------------------------------
// Pre-pack W1 (fp32) into WMMA-fragment-swizzled bf16 hi/lo arrays.
// One thread per (t,ks,cb,lane) writes its 16-element fragment slice (32B).
// ---------------------------------------------------------------------------
__global__ void __launch_bounds__(256)
mhc_pack(const float* __restrict__ W1, __bf16* __restrict__ whi,
         __bf16* __restrict__ wlo)
{
    int gid  = blockIdx.x * 256 + threadIdx.x;      // 131072 threads
    int lane = gid & 31;
    int cb   = (gid >> 5) & 7;
    int ks   = (gid >> 8) & 31;
    int t    = gid >> 13;

    int col  = cb * 16 + (lane & 15);
    int half = lane >> 4;

    bf16x16 vh, vl;
#pragma unroll
    for (int j = 0; j < 16; ++j) {
        int k = ks * 32 + kmap(j, half);
        float w = W1[((size_t)t * DN + k) * HN + col];
        __bf16 h = (__bf16)w;
        vh[j] = h;
        vl[j] = (__bf16)(w - (float)h);
    }
    *(bf16x16*)(whi + (size_t)gid * 16) = vh;
    *(bf16x16*)(wlo + (size_t)gid * 16) = vl;
}

// ---------------------------------------------------------------------------
// Grouped GEMM: 32-row tile per workgroup; 8 wave32s, wave w owns H columns
// [16w,16w+16) for both 16-row halves. Inner loop per K-step of 32:
// 8 ds_load_b128 (A hi/lo, two row groups), 4 global_load_b128 (packed B
// hi/lo), 6 interleaved WMMAs. x staging is register-pipelined.
// ---------------------------------------------------------------------------
__global__ void __launch_bounds__(256)
mhc_gemm(const float* __restrict__ x,
         const __bf16* __restrict__ whi,
         const __bf16* __restrict__ wlo,
         const float* __restrict__ b1,
         const float* __restrict__ W2,
         const float* __restrict__ b2,
         const int*   __restrict__ ws,
         float*       __restrict__ out)
{
    __shared__ __bf16 s_ahi[MTILE][136];  // +8 pad: conflict-free frag reads
    __shared__ __bf16 s_alo[MTILE][136];
    __shared__ float  s_h[MTILE][132];    // layer-1 output tile (relu'd)
    __shared__ int    s_idx[MTILE];

    const int total = ws[WS_OFF + TN];
    const int rowBase = blockIdx.x * MTILE;
    if (rowBase >= total) return;         // uniform: EXEC stays full

    int t = 0;                            // offsets are 32-aligned
#pragma unroll
    for (int i = 1; i < TN; ++i)
        if (rowBase >= ws[WS_OFF + i]) t = i;

    const int tid  = threadIdx.x;
    const int lane = tid & 31;
    const int wv   = tid >> 5;            // 0..7 -> H column block
    const int half = lane >> 4;
    const int ln   = lane & 15;
    const int colBase = wv * 16;

    if (tid < MTILE) s_idx[tid] = ws[WS_BUCK + rowBase + tid];
    __syncthreads();

    // packed-W1 fragment pointers for this (t, colblk, lane)
    const __bf16* __restrict__ wph =
        whi + ((size_t)t * 32 * 8 + wv) * 32 * 16 + (size_t)lane * 16;
    const __bf16* __restrict__ wpl =
        wlo + ((size_t)t * 32 * 8 + wv) * 32 * 16 + (size_t)lane * 16;
    const size_t KS_STRIDE = 8 * 32 * 16;   // elements per ks step

    f32x8 acc0 = {};
    f32x8 acc1 = {};

    // staging: thread handles row tid>>3, cols (tid&7)*16 .. +15
    const int sr = tid >> 3;
    const int li = tid & 7;
    int srow = s_idx[sr];
    const f32x4* __restrict__ xsrc = (const f32x4*)
        (x + (size_t)(srow < 0 ? 0 : srow) * DN + li * 16);

    // prologue: prefetch chunk 0 (non-temporal: one-touch 256MB stream)
    f32x4 r0 = __builtin_nontemporal_load(xsrc + 0);
    f32x4 r1 = __builtin_nontemporal_load(xsrc + 1);
    f32x4 r2 = __builtin_nontemporal_load(xsrc + 2);
    f32x4 r3 = __builtin_nontemporal_load(xsrc + 3);

    int kks = 0;                           // global ks index 0..31
    for (int kb = 0; kb < DN; kb += 128) {
        __syncthreads();                   // prior fragment reads done
        // ---- convert prefetched regs, store chunk to LDS as bf16 hi/lo ----
        {
            float fv[16] = {r0[0], r0[1], r0[2], r0[3],
                            r1[0], r1[1], r1[2], r1[3],
                            r2[0], r2[1], r2[2], r2[3],
                            r3[0], r3[1], r3[2], r3[3]};
#pragma unroll
            for (int q = 0; q < 16; ++q) {
                float f = fv[q];
                __bf16 h = (__bf16)f;
                s_ahi[sr][li * 16 + q] = h;
                s_alo[sr][li * 16 + q] = (__bf16)(f - (float)h);
            }
        }
        __syncthreads();

        // ---- prefetch next chunk while this chunk computes ----
        if (kb + 128 < DN) {
            const f32x4* p4 = xsrc + ((kb + 128) >> 2);
            r0 = __builtin_nontemporal_load(p4 + 0);
            r1 = __builtin_nontemporal_load(p4 + 1);
            r2 = __builtin_nontemporal_load(p4 + 2);
            r3 = __builtin_nontemporal_load(p4 + 3);
        }

        // ---- 4 K-steps of 32 ----
#pragma unroll
        for (int ks = 0; ks < 128; ks += 32) {
            bf16x16 ah0, al0, ah1, al1;
#pragma unroll
            for (int j = 0; j < 16; ++j) {
                int k = ks + kmap(j, half);
                ah0[j] = s_ahi[ln][k];
                al0[j] = s_alo[ln][k];
                ah1[j] = s_ahi[16 + ln][k];
                al1[j] = s_alo[16 + ln][k];
            }
            bf16x16 bh = *(const bf16x16*)(wph + (size_t)kks * KS_STRIDE);
            bf16x16 bl = *(const bf16x16*)(wpl + (size_t)kks * KS_STRIDE);
            ++kks;

            // interleave the two accumulators to break RAW chains
            acc0 = __builtin_amdgcn_wmma_f32_16x16x32_bf16(
                       false, ah0, false, bh, (short)0, acc0, false, false);
            acc1 = __builtin_amdgcn_wmma_f32_16x16x32_bf16(
                       false, ah1, false, bh, (short)0, acc1, false, false);
            acc0 = __builtin_amdgcn_wmma_f32_16x16x32_bf16(
                       false, ah0, false, bl, (short)0, acc0, false, false);
            acc1 = __builtin_amdgcn_wmma_f32_16x16x32_bf16(
                       false, ah1, false, bl, (short)0, acc1, false, false);
            acc0 = __builtin_amdgcn_wmma_f32_16x16x32_bf16(
                       false, al0, false, bh, (short)0, acc0, false, false);
            acc1 = __builtin_amdgcn_wmma_f32_16x16x32_bf16(
                       false, al1, false, bh, (short)0, acc1, false, false);
        }
    }

    // ---- bias + relu, write h tile to LDS ----
    // C/D layout: element r -> row m = r + 8*half, col n = lane&15
    {
        float bias = b1[t * HN + colBase + ln];
#pragma unroll
        for (int r = 0; r < 8; ++r) {
            int m = r + (half ? 8 : 0);
            float v0 = acc0[r] + bias;
            float v1 = acc1[r] + bias;
            s_h[m][colBase + ln]      = v0 > 0.0f ? v0 : 0.0f;
            s_h[16 + m][colBase + ln] = v1 > 0.0f ? v1 : 0.0f;
        }
    }
    __syncthreads();

    // ---- layer 2: [32x128] @ [128x10] + b2, scatter to out ----
    if (tid < 16 * CN) {
        int r = tid / CN;
        int c = tid - r * CN;
        const float* __restrict__ W2t = W2 + (size_t)t * HN * CN + c;
        float bias2 = b2[t * CN + c];
        float s0 = 0.0f, s1 = 0.0f;
#pragma unroll 8
        for (int k = 0; k < HN; ++k) {
            float w = W2t[(size_t)k * CN];
            s0 += s_h[r][k] * w;
            s1 += s_h[16 + r][k] * w;
        }
        int row0 = s_idx[r];
        int row1 = s_idx[16 + r];
        if (row0 >= 0)
            __builtin_nontemporal_store(s0 + bias2, out + (size_t)row0 * CN + c);
        if (row1 >= 0)
            __builtin_nontemporal_store(s1 + bias2, out + (size_t)row1 * CN + c);
    }
}

// ---------------------------------------------------------------------------
extern "C" void kernel_launch(void* const* d_in, const int* in_sizes, int n_in,
                              void* d_out, int out_size, void* d_ws, size_t ws_size,
                              hipStream_t stream) {
    const float* x    = (const float*)d_in[0];
    const int*   task = (const int*)  d_in[1];
    const float* W1   = (const float*)d_in[2];
    const float* b1   = (const float*)d_in[3];
    const float* W2   = (const float*)d_in[4];
    const float* b2   = (const float*)d_in[5];
    float* out = (float*)d_out;
    int*   ws  = (int*)d_ws;

    size_t packOff = ((size_t)WS_INTS * sizeof(int) + 255) & ~(size_t)255;
    __bf16* whi = (__bf16*)((char*)d_ws + packOff);
    __bf16* wlo = whi + WPACK_ELEMS;

    mhc_init<<<1, 64, 0, stream>>>(ws);
    mhc_hist<<<BN / 256, 256, 0, stream>>>(task, ws);
    mhc_offsets<<<1, 1, 0, stream>>>(ws);
    mhc_fill<<<(BUCK_CAP + 255) / 256, 256, 0, stream>>>(ws);
    mhc_scatter<<<BN / 256, 256, 0, stream>>>(task, ws);
    mhc_pack<<<(TN * 32 * 8 * 32) / 256, 256, 0, stream>>>(W1, whi, wlo);
    mhc_gemm<<<NTILES, 256, 0, stream>>>(x, whi, wlo, b1, W2, b2, ws, out);
}
